// CustomMoE_16312285790685
// MI455X (gfx1250) — matrix operand order repaired and checked by reading, not compiled, not run on hardware
//
#include <hip/hip_runtime.h>

// ---------------------------------------------------------------------------
// MoE forward for MI455X (gfx1250, wave32, WMMA).
// Compute-bound (~344 GFLOP vs ~35us of HBM time at 23.3 TB/s): expert GEMMs
// use v_wmma_f32_16x16x32_bf16 with f32 accumulation; operands converted
// f32->bf16 while staging into LDS. h kept in bf16. gemm2 A-tiles use
// global_load_async_to_lds_b128 (ASYNCcnt) since they are straight bf16 copies.
// ---------------------------------------------------------------------------

typedef __attribute__((ext_vector_type(16))) __bf16 v16bf;
typedef __attribute__((ext_vector_type(8)))  __bf16 v8bf;
typedef __attribute__((ext_vector_type(4)))  __bf16 v4bf;
typedef __attribute__((ext_vector_type(8)))  float  v8f;

namespace {
constexpr int T   = 8192;
constexpr int D   = 1024;
constexpr int H   = 4096;
constexpr int E   = 16;
constexpr int CAP = 1280;
constexpr int NFLAT = 2 * T;           // top-2, slot-major
constexpr int GATE_BLOCKS = T / 256;   // 32
}

// ---------------------------------------------------------------- init ------
__global__ void moe_init(int* ce_cnt, float* laux) {
  const int tid = threadIdx.x;
  if (tid < 16) ce_cnt[tid] = 0;
  if (tid == 16) *laux = 0.f;
}

// -------------------------------------------------------------- gating ------
__global__ __launch_bounds__(256) void moe_gating(
    const float* __restrict__ x, const float* __restrict__ wg,
    int* __restrict__ topk_idx, float* __restrict__ topk_gate,
    float* __restrict__ me_part, int* __restrict__ ce_cnt) {
  __shared__ float red[256 * 16];
  const int tid = threadIdx.x;
  const int t = blockIdx.x * 256 + tid;

  float acc[16];
#pragma unroll
  for (int e = 0; e < 16; ++e) acc[e] = 0.f;
  const float* xr = x + (size_t)t * D;
  for (int d = 0; d < D; ++d) {
    const float xv = xr[d];
    const float* w = wg + d * 16;           // wg is [D][E], broadcast per wave
#pragma unroll
    for (int e = 0; e < 16; ++e) acc[e] += xv * w[e];
  }
  // softmax over 16 experts
  float m = acc[0];
#pragma unroll
  for (int e = 1; e < 16; ++e) m = fmaxf(m, acc[e]);
  float sum = 0.f;
#pragma unroll
  for (int e = 0; e < 16; ++e) { acc[e] = __expf(acc[e] - m); sum += acc[e]; }
  const float inv = 1.f / sum;
#pragma unroll
  for (int e = 0; e < 16; ++e) acc[e] *= inv;
  // top-2, lowest index wins ties (matches lax.top_k)
  int e0 = 0; float g0 = acc[0];
#pragma unroll
  for (int e = 1; e < 16; ++e) if (acc[e] > g0) { g0 = acc[e]; e0 = e; }
  int e1 = 0; float g1 = -1.f;
#pragma unroll
  for (int e = 0; e < 16; ++e) if (e != e0 && acc[e] > g1) { g1 = acc[e]; e1 = e; }
  const float denom = g0 + g1;
  topk_idx[t]      = e0;
  topk_idx[T + t]  = e1;
  topk_gate[t]     = g0 / denom;
  topk_gate[T + t] = g1 / denom;
  atomicAdd(&ce_cnt[e0], 1);               // integer atomic: deterministic

  // deterministic per-block tree reduction of softmax scores (for me / l_aux)
#pragma unroll
  for (int e = 0; e < 16; ++e) red[tid * 16 + e] = acc[e];
  __syncthreads();
  for (int off = 128; off > 0; off >>= 1) {
    if (tid < off) {
#pragma unroll
      for (int e = 0; e < 16; ++e) red[tid * 16 + e] += red[(tid + off) * 16 + e];
    }
    __syncthreads();
  }
  if (tid < 16) me_part[blockIdx.x * 16 + tid] = red[tid];
}

// ---------------------------------------------------- capacity scan ---------
__global__ __launch_bounds__(1024) void moe_scan(
    const int* __restrict__ topk_idx, const float* __restrict__ topk_gate,
    int* __restrict__ slot_tok, int* __restrict__ comb_pos,
    float* __restrict__ comb_gate, const float* __restrict__ me_part,
    const int* __restrict__ ce_cnt, float* __restrict__ laux) {
  __shared__ int ss[1024];
  const int tid = threadIdx.x;

  for (int i = tid; i < E * CAP; i += 1024) slot_tok[i] = -1;

  if (tid == 0) {   // l_aux (tiny, sequential -> deterministic)
    float l = 0.f;
    for (int e = 0; e < 16; ++e) {
      float me = 0.f;
      for (int b = 0; b < GATE_BLOCKS; ++b) me += me_part[b * 16 + e];
      me /= (float)T;
      l += me * ((float)ce_cnt[e] / (float)T);
    }
    *laux = l * (float)E;
  }

  const int per = NFLAT / 1024;   // 16 items per thread, contiguous, in order
  const int base_f = tid * per;
  int cnt[16];
#pragma unroll
  for (int e = 0; e < 16; ++e) cnt[e] = 0;
  for (int i = 0; i < per; ++i) cnt[topk_idx[base_f + i]]++;

  int ebase[16];
  for (int e = 0; e < 16; ++e) {
    __syncthreads();
    ss[tid] = cnt[e];
    __syncthreads();
    for (int off = 1; off < 1024; off <<= 1) {     // Hillis-Steele inclusive
      const int v = (tid >= off) ? ss[tid - off] : 0;
      __syncthreads();
      ss[tid] += v;
      __syncthreads();
    }
    ebase[e] = (tid == 0) ? 0 : ss[tid - 1];       // exclusive prefix
  }

  for (int i = 0; i < per; ++i) {
    const int f = base_f + i;
    const int e = topk_idx[f];
    const int p = ebase[e]++;                      // counts even past CAP
    const bool keep = p < CAP;
    comb_pos[f]  = keep ? p : (CAP - 1);
    comb_gate[f] = keep ? topk_gate[f] : 0.f;
    if (keep) slot_tok[e * CAP + p] = f & (T - 1); // token id = f % T
  }
}

// --------------------------------------------------- WMMA fragment IO -------
__device__ inline v16bf load_a_frag(const __bf16* rowp, int laneHi) {
  // A 16x32 bf16: lane<16 holds K=0..15 of its row, lane>=16 holds K=16..31
  return *(const v16bf*)(rowp + laneHi * 16);
}
__device__ inline v16bf load_b_frag(const __bf16* colp, int laneHi) {
  // B 32x16 bf16: lane<16 holds K=0..15 of its column, lane>=16 holds K=16..31
  return *(const v16bf*)(colp + laneHi * 16);
}

// 16B global -> LDS async copy (ASYNCcnt). LDS offset = low 32 bits of the
// generic pointer (LDS aperture keeps the wave-relative byte address there).
__device__ inline void async_copy16(const void* gptr, void* lptr) {
  const unsigned lds = (unsigned)(unsigned long long)(uintptr_t)lptr;
  asm volatile("global_load_async_to_lds_b128 %0, %1, off"
               :: "v"(lds), "v"(gptr) : "memory");
}
__device__ inline void async_wait0() {
  asm volatile("s_wait_asynccnt 0x0" ::: "memory");
}

// --------------------------------------------------------------- GEMM1 ------
// h[e,row,col] = relu(disp[e,row,:] @ fc1_w[e,:,col] + fc1_b), h stored bf16.
// Block = 256 threads (8 waves), tile 128x128, K-step 32. Wave tile 32x64.
__global__ __launch_bounds__(256) void moe_gemm1(
    const float* __restrict__ x, const float* __restrict__ fc1w,
    const float* __restrict__ fc1b, const int* __restrict__ slot_tok,
    __bf16* __restrict__ h) {
  __shared__ __bf16 As[128 * 32];
  __shared__ __bf16 Bs[128 * 32];
  __shared__ int toks[128];
  const int tid = threadIdx.x;
  const int nblk = blockIdx.x, e = blockIdx.y, mblk = blockIdx.z;
  const int lane = tid & 31, wave = tid >> 5;
  const int lane15 = lane & 15, laneHi = lane >> 4;
  const int wm = (wave & 3) * 32, wn = (wave >> 2) * 64;

  if (tid < 128) toks[tid] = slot_tok[e * CAP + mblk * 128 + tid];
  __syncthreads();

  // per-thread gather state: 4 fixed rows, branch-free via clamp + 0/1 scale
  const float* rp[4];
  float sc[4];
#pragma unroll
  for (int i = 0; i < 4; ++i) {
    const int r = i * 32 + (tid >> 3);
    const int tok = toks[r];
    rp[i] = x + (size_t)(tok < 0 ? 0 : tok) * D;
    sc[i] = (tok < 0) ? 0.f : 1.f;
  }
  const int kk4 = (tid & 7) * 4;        // A: 4 consecutive f32 per thread/row
  const int bcol = tid >> 1;            // B: 2 threads per column
  const int bk0 = (tid & 1) * 16;

  v8f acc[2][4];
#pragma unroll
  for (int mi = 0; mi < 2; ++mi)
#pragma unroll
    for (int ni = 0; ni < 4; ++ni)
#pragma unroll
      for (int v = 0; v < 8; ++v) acc[mi][ni][v] = 0.f;

  const float* wbase = fc1w + (size_t)e * D * H + (size_t)nblk * 128;
  for (int k0 = 0; k0 < D; k0 += 32) {
    // stage A: unconditional vector loads, f32 -> bf16
#pragma unroll
    for (int i = 0; i < 4; ++i) {
      const int r = i * 32 + (tid >> 3);
      const float4 v = *(const float4*)(rp[i] + k0 + kk4);
      v4bf b;
      b[0] = (__bf16)(v.x * sc[i]); b[1] = (__bf16)(v.y * sc[i]);
      b[2] = (__bf16)(v.z * sc[i]); b[3] = (__bf16)(v.w * sc[i]);
      *(v4bf*)&As[r * 32 + kk4] = b;
    }
    // stage B: 16 strided loads -> two 16B LDS stores (column-major in LDS)
    {
      const float* wc = wbase + (size_t)(k0 + bk0) * H + bcol;
      v8bf p0, p1;
#pragma unroll
      for (int kk = 0; kk < 8; ++kk) p0[kk] = (__bf16)wc[(size_t)kk * H];
#pragma unroll
      for (int kk = 0; kk < 8; ++kk) p1[kk] = (__bf16)wc[(size_t)(kk + 8) * H];
      *(v8bf*)&Bs[bcol * 32 + bk0] = p0;
      *(v8bf*)&Bs[bcol * 32 + bk0 + 8] = p1;
      if (k0 + 32 < D)
        __builtin_prefetch(wbase + (size_t)(k0 + 32 + bk0) * H + bcol, 0, 1);
    }
    __syncthreads();

    v16bf af[2], bfr[4];
#pragma unroll
    for (int mi = 0; mi < 2; ++mi)
      af[mi] = load_a_frag(&As[(wm + mi * 16 + lane15) * 32], laneHi);
#pragma unroll
    for (int ni = 0; ni < 4; ++ni)
      bfr[ni] = load_b_frag(&Bs[(wn + ni * 16 + lane15) * 32], laneHi);
#pragma unroll
    for (int mi = 0; mi < 2; ++mi)
#pragma unroll
      for (int ni = 0; ni < 4; ++ni)
        acc[mi][ni] = __builtin_amdgcn_wmma_f32_16x16x32_bf16(
            false, af[mi], false, bfr[ni], (short)0, acc[mi][ni], false, false);
    __syncthreads();
  }

  // epilogue: bias + relu, store bf16
#pragma unroll
  for (int mi = 0; mi < 2; ++mi)
#pragma unroll
    for (int ni = 0; ni < 4; ++ni) {
      const int col = nblk * 128 + wn + ni * 16 + lane15;
      const float b = fc1b[(size_t)e * H + col];
#pragma unroll
      for (int v = 0; v < 8; ++v) {
        const int row = mblk * 128 + wm + mi * 16 + v + laneHi * 8;
        float val = acc[mi][ni][v] + b;
        val = val > 0.f ? val : 0.f;
        h[((size_t)e * CAP + row) * H + col] = (__bf16)val;
      }
    }
}

// --------------------------------------------------------------- GEMM2 ------
// y[e,row,col] = h[e,row,:] @ fc2_w[e,:,col] + fc2_b, y stored f32.
__global__ __launch_bounds__(256) void moe_gemm2(
    const __bf16* __restrict__ h, const float* __restrict__ fc2w,
    const float* __restrict__ fc2b, float* __restrict__ y) {
  __shared__ __bf16 As[128 * 32];
  __shared__ __bf16 Bs[128 * 32];
  const int tid = threadIdx.x;
  const int nblk = blockIdx.x, e = blockIdx.y, mblk = blockIdx.z;
  const int lane = tid & 31, wave = tid >> 5;
  const int lane15 = lane & 15, laneHi = lane >> 4;
  const int wm = (wave & 3) * 32, wn = (wave >> 2) * 64;
  const int bcol = tid >> 1;
  const int bk0 = (tid & 1) * 16;

  v8f acc[2][4];
#pragma unroll
  for (int mi = 0; mi < 2; ++mi)
#pragma unroll
    for (int ni = 0; ni < 4; ++ni)
#pragma unroll
      for (int v = 0; v < 8; ++v) acc[mi][ni][v] = 0.f;

  const __bf16* hb = h + ((size_t)e * CAP + (size_t)mblk * 128) * H;
  const float* wbase = fc2w + (size_t)e * H * D + (size_t)nblk * 128;
  for (int k0 = 0; k0 < H; k0 += 32) {
    // stage A: bf16 tile copy via async global->LDS (ASYNCcnt), 16B per lane
#pragma unroll
    for (int c2 = 0; c2 < 2; ++c2) {
      const int ch = c2 * 256 + tid;        // 512 chunks of 8 bf16
      const int r = ch >> 2, kk = (ch & 3) * 8;
      async_copy16(hb + (size_t)r * H + k0 + kk, &As[r * 32 + kk]);
    }
    // stage B: f32 -> bf16
    {
      const float* wc = wbase + (size_t)(k0 + bk0) * D + bcol;
      v8bf p0, p1;
#pragma unroll
      for (int kk = 0; kk < 8; ++kk) p0[kk] = (__bf16)wc[(size_t)kk * D];
#pragma unroll
      for (int kk = 0; kk < 8; ++kk) p1[kk] = (__bf16)wc[(size_t)(kk + 8) * D];
      *(v8bf*)&Bs[bcol * 32 + bk0] = p0;
      *(v8bf*)&Bs[bcol * 32 + bk0 + 8] = p1;
      if (k0 + 32 < H)
        __builtin_prefetch(wbase + (size_t)(k0 + 32 + bk0) * D + bcol, 0, 1);
    }
    async_wait0();
    __syncthreads();

    v16bf af[2], bfr[4];
#pragma unroll
    for (int mi = 0; mi < 2; ++mi)
      af[mi] = load_a_frag(&As[(wm + mi * 16 + lane15) * 32], laneHi);
#pragma unroll
    for (int ni = 0; ni < 4; ++ni)
      bfr[ni] = load_b_frag(&Bs[(wn + ni * 16 + lane15) * 32], laneHi);
#pragma unroll
    for (int mi = 0; mi < 2; ++mi)
#pragma unroll
      for (int ni = 0; ni < 4; ++ni)
        acc[mi][ni] = __builtin_amdgcn_wmma_f32_16x16x32_bf16(
            false, af[mi], false, bfr[ni], (short)0, acc[mi][ni], false, false);
    __syncthreads();
  }

#pragma unroll
  for (int mi = 0; mi < 2; ++mi)
#pragma unroll
    for (int ni = 0; ni < 4; ++ni) {
      const int col = nblk * 128 + wn + ni * 16 + lane15;
      const float b = fc2b[(size_t)e * D + col];
#pragma unroll
      for (int v = 0; v < 8; ++v) {
        const int row = mblk * 128 + wm + mi * 16 + v + laneHi * 8;
        y[((size_t)e * CAP + row) * D + col] = acc[mi][ni][v] + b;
      }
    }
}

// ------------------------------------------------------------- combine ------
__global__ __launch_bounds__(128) void moe_combine(
    const float* __restrict__ y, const int* __restrict__ topk_idx,
    const int* __restrict__ comb_pos, const float* __restrict__ comb_gate,
    const float* __restrict__ laux, float* __restrict__ out) {
  const int t = blockIdx.x, tid = threadIdx.x;
  const int e0 = topk_idx[t], e1 = topk_idx[T + t];
  const int p0 = comb_pos[t], p1 = comb_pos[T + t];
  const float g0 = comb_gate[t], g1 = comb_gate[T + t];
  const float4* r0 = (const float4*)(y + ((size_t)e0 * CAP + p0) * D);
  const float4* r1 = (const float4*)(y + ((size_t)e1 * CAP + p1) * D);
  float4* o = (float4*)(out + (size_t)t * D);
#pragma unroll
  for (int i = 0; i < 2; ++i) {
    const int c = i * 128 + tid;          // D/4 = 256 float4 per row
    const float4 a = r0[c], b = r1[c];
    float4 v;
    v.x = g0 * a.x + g1 * b.x;
    v.y = g0 * a.y + g1 * b.y;
    v.z = g0 * a.z + g1 * b.z;
    v.w = g0 * a.w + g1 * b.w;
    o[c] = v;
  }
  if (t == 0 && tid == 0) out[(size_t)T * D] = *laux;   // l_aux appended
}

// --------------------------------------------------------------- launch -----
extern "C" void kernel_launch(void* const* d_in, const int* in_sizes, int n_in,
                              void* d_out, int out_size, void* d_ws, size_t ws_size,
                              hipStream_t stream) {
  (void)in_sizes; (void)n_in; (void)out_size; (void)ws_size;
  const float* x    = (const float*)d_in[0];
  const float* wg   = (const float*)d_in[1];
  const float* fc1w = (const float*)d_in[2];
  const float* fc1b = (const float*)d_in[3];
  const float* fc2w = (const float*)d_in[4];
  const float* fc2b = (const float*)d_in[5];
  float* out = (float*)d_out;
  char* ws = (char*)d_ws;

  auto aln = [](size_t v) { return (v + 255) & ~(size_t)255; };
  size_t off = 0;
  const size_t OFF_H    = off; off = aln(off + (size_t)E * CAP * H * 2);  // h bf16
  const size_t OFF_Y    = off; off = aln(off + (size_t)E * CAP * D * 4);  // y f32
  const size_t OFF_SLOT = off; off = aln(off + (size_t)E * CAP * 4);
  const size_t OFF_TI   = off; off = aln(off + (size_t)NFLAT * 4);
  const size_t OFF_TG   = off; off = aln(off + (size_t)NFLAT * 4);
  const size_t OFF_CP   = off; off = aln(off + (size_t)NFLAT * 4);
  const size_t OFF_CG   = off; off = aln(off + (size_t)NFLAT * 4);
  const size_t OFF_ME   = off; off = aln(off + (size_t)GATE_BLOCKS * 16 * 4);
  const size_t OFF_CE   = off; off = aln(off + 64);
  const size_t OFF_LX   = off; off = aln(off + 4);

  __bf16* h_ws    = (__bf16*)(ws + OFF_H);
  float*  y_ws    = (float*)(ws + OFF_Y);
  int*    slot    = (int*)(ws + OFF_SLOT);
  int*    tki     = (int*)(ws + OFF_TI);
  float*  tkg     = (float*)(ws + OFF_TG);
  int*    cpos    = (int*)(ws + OFF_CP);
  float*  cgate   = (float*)(ws + OFF_CG);
  float*  me_part = (float*)(ws + OFF_ME);
  int*    ce_cnt  = (int*)(ws + OFF_CE);
  float*  laux    = (float*)(ws + OFF_LX);

  moe_init<<<1, 32, 0, stream>>>(ce_cnt, laux);
  moe_gating<<<GATE_BLOCKS, 256, 0, stream>>>(x, wg, tki, tkg, me_part, ce_cnt);
  moe_scan<<<1, 1024, 0, stream>>>(tki, tkg, slot, cpos, cgate, me_part, ce_cnt, laux);
  moe_gemm1<<<dim3(H / 128, E, CAP / 128), 256, 0, stream>>>(x, fc1w, fc1b, slot, h_ws);
  moe_gemm2<<<dim3(D / 128, E, CAP / 128), 256, 0, stream>>>(h_ws, fc2w, fc2b, y_ws);
  moe_combine<<<T, 128, 0, stream>>>(y_ws, tki, cpos, cgate, laux, out);
}
